// MockTriangleAttention_75282186764816
// MI455X (gfx1250) — matrix operand order repaired and checked by reading, not compiled, not run on hardware
//
#include <hip/hip_runtime.h>
#include <stdint.h>

// ---------------- CDNA5 WMMA types ----------------
typedef __attribute__((ext_vector_type(16))) __bf16 v16bf;
typedef __attribute__((ext_vector_type(2)))  __bf16 v2bf;
typedef __attribute__((ext_vector_type(8)))  float  v8f;

#define J_DIM   384
#define C_DIM   128
#define N_HEADS 4
#define D_HEAD  32
#define INF_F   1000000000.0f
#define EPS_F   1e-5f

// LDS layout (bytes), all bf16 payloads, XOR-swizzled
#define XN_OFF  0          // [384][128] bf16 = 98304
#define K_OFF   98304      // [384][128] bf16 = 98304
#define VT_OFF  196608     // [128][384] bf16 = 98304
#define STG_OFF 294912     // 8 waves * 1024B staging
#define SMEM_BYTES 303104  // 296 KB  (<= 320 KB WGP LDS)

// ---------------- helpers ----------------
__device__ __forceinline__ uint32_t rm_addr(uint32_t row, uint32_t colByte) {
  // row-major [*][128] bf16 (256B rows), xor-swizzle 16B chunks by row
  return row * 256u + (colByte ^ ((row & 15u) << 4));
}
__device__ __forceinline__ uint32_t vt_addr(uint32_t ch, uint32_t posByte) {
  // [128][384] bf16 (768B rows), swizzle within 256B sub-blocks
  return ch * 768u + (posByte ^ ((ch & 15u) << 4));
}
__device__ __forceinline__ uint32_t st_addr(uint32_t row, uint32_t colByte) {
  // per-wave staging [16][32] bf16 (64B rows), rotate 16B chunk by row>>2
  uint32_t chunk = ((colByte >> 4) + (row >> 2)) & 3u;
  return row * 64u + (chunk << 4) + (colByte & 15u);
}
// native bf16 converts: let clang emit v_cvt_pk_bf16_f32 / v_cvt_bf16
__device__ __forceinline__ uint16_t f2bf(float f) {
  union { __bf16 b; uint16_t u; } cv;
  cv.b = (__bf16)f;
  return cv.u;
}
__device__ __forceinline__ uint32_t pack2bf(float a, float b) {
  union { v2bf v; uint32_t u; } cv;
  cv.v = v2bf{(__bf16)a, (__bf16)b};
  return cv.u;
}
__device__ __forceinline__ v16bf ld_frag(const void* p0, const void* p1) {
  union { v16bf v; uint4 q[2]; } f;
  f.q[0] = *(const uint4*)p0;
  f.q[1] = *(const uint4*)p1;
  return f.v;
}
__device__ __forceinline__ v8f vzero() {
  v8f z = {0.f,0.f,0.f,0.f,0.f,0.f,0.f,0.f};
  return z;
}
__device__ __forceinline__ v8f wmma_bf16(v16bf a, v16bf b, v8f c) {
  return __builtin_amdgcn_wmma_f32_16x16x32_bf16(false, a, false, b, (short)0, c, false, false);
}
__device__ __forceinline__ float redmax16(float v) {
  v = fmaxf(v, __shfl_xor(v, 1));
  v = fmaxf(v, __shfl_xor(v, 2));
  v = fmaxf(v, __shfl_xor(v, 4));
  v = fmaxf(v, __shfl_xor(v, 8));
  return v;
}
__device__ __forceinline__ float redsum16(float v) {
  v += __shfl_xor(v, 1);
  v += __shfl_xor(v, 2);
  v += __shfl_xor(v, 4);
  v += __shfl_xor(v, 8);
  return v;
}

// ---------------- weight prep: fp32 [cin][cout] -> bf16 W^T [cout][cin] ----------------
__global__ __launch_bounds__(256) void ta_prep_weights(
    const float* __restrict__ wq, const float* __restrict__ wk,
    const float* __restrict__ wv, const float* __restrict__ wo,
    uint16_t* __restrict__ wsb) {
  uint32_t idx = blockIdx.x * 256u + threadIdx.x;       // 0..65535
  uint32_t m = idx >> 14;                               // which matrix
  uint32_t r = (idx >> 7) & 127u;                       // cout (row of W^T)
  uint32_t c = idx & 127u;                              // cin  (col of W^T)
  const float* src = (m == 0) ? wq : (m == 1) ? wk : (m == 2) ? wv : wo;
  wsb[idx] = f2bf(src[c * 128u + r]);                   // transpose
}

// ---------------- fused triangle-attention kernel, one block per i ----------------
__global__ __launch_bounds__(256, 1) void ta_fused_kernel(
    const float* __restrict__ x, const float* __restrict__ mask,
    const float* __restrict__ ln_w, const float* __restrict__ ln_b,
    const float* __restrict__ b_o, const uint16_t* __restrict__ wsw,
    float* __restrict__ out) {
  extern __shared__ char smem[];
  const int i    = blockIdx.x;
  // wave index is wave-uniform: force it into an SGPR so phase loops /
  // branches are scalar control flow (keeps EXEC all-ones for WMMA regions)
  const int wave = __builtin_amdgcn_readfirstlane(threadIdx.x >> 5);
  const int lane = threadIdx.x & 31;
  const int half = lane >> 4;        // 0: lanes 0-15, 1: lanes 16-31
  const int l16  = lane & 15;

  const uint16_t* wqT = wsw;
  const uint16_t* wkT = wsw + 16384;
  const uint16_t* wvT = wsw + 32768;
  const uint16_t* woT = wsw + 49152;

  // ---------------- Phase A: LayerNorm -> XN (bf16, LDS) ----------------
  {
    const float* xrow = x + (size_t)i * J_DIM * C_DIM;
    float lw0 = ln_w[lane * 4 + 0], lw1 = ln_w[lane * 4 + 1];
    float lw2 = ln_w[lane * 4 + 2], lw3 = ln_w[lane * 4 + 3];
    float lb0 = ln_b[lane * 4 + 0], lb1 = ln_b[lane * 4 + 1];
    float lb2 = ln_b[lane * 4 + 2], lb3 = ln_b[lane * 4 + 3];
    for (int j = wave; j < J_DIM; j += 8) {
      // prefetch next row chunk for this wave (global_prefetch_b8)
      if (j + 8 < J_DIM)
        __builtin_prefetch(xrow + (j + 8) * C_DIM + lane * 4, 0, 3);
      float4 v = *(const float4*)(xrow + j * C_DIM + lane * 4);
      float s  = v.x + v.y + v.z + v.w;
      float s2 = v.x*v.x + v.y*v.y + v.z*v.z + v.w*v.w;
      #pragma unroll
      for (int m = 1; m <= 16; m <<= 1) { s += __shfl_xor(s, m); s2 += __shfl_xor(s2, m); }
      float mu  = s * (1.0f / 128.0f);
      float var = s2 * (1.0f / 128.0f) - mu * mu;
      float rs  = rsqrtf(var + EPS_F);
      uint2 st;
      st.x = pack2bf((v.x - mu) * rs * lw0 + lb0, (v.y - mu) * rs * lw1 + lb1);
      st.y = pack2bf((v.z - mu) * rs * lw2 + lb2, (v.w - mu) * rs * lw3 + lb3);
      *(uint2*)(smem + XN_OFF + rm_addr(j, lane * 8)) = st;
    }
  }
  __syncthreads();

  // ---------------- Phase B: K = XN@Wk (row-major), V^T = (XN@Wv)^T ----------------
  // 384 tile jobs: mat(2) x rowtile(24) x ntile(8); all indices wave-uniform (SALU)
  for (int t = 0; t < 48; ++t) {
    int jid = wave + t * 8;
    int mat = jid / 192;
    int rem = jid - mat * 192;
    int rowtile = rem >> 3;
    int ntile   = rem & 7;
    const uint16_t* wT = mat ? wvT : wkT;
    int arow = rowtile * 16 + l16;
    int cout = ntile * 16 + l16;
    v8f acc = vzero();
    #pragma unroll
    for (int kc = 0; kc < 4; ++kc) {
      v16bf a = ld_frag(smem + XN_OFF + rm_addr(arow, kc * 64 + half * 32),
                        smem + XN_OFF + rm_addr(arow, kc * 64 + half * 32 + 16));
      const char* wb = (const char*)wT + ((size_t)cout * 128 + kc * 32 + half * 16) * 2;
      v16bf b = ld_frag(wb, wb + 16);
      acc = wmma_bf16(a, b, acc);
    }
    if (mat == 0) {  // K row-major [pos][chan]
      int rbase = rowtile * 16 + half * 8;
      #pragma unroll
      for (int r = 0; r < 8; ++r)
        *(uint16_t*)(smem + K_OFF + rm_addr(rbase + r, cout * 2)) = f2bf(acc[r]);
    } else {         // V transposed [chan][pos]
      int pbase = rowtile * 16 + half * 8;
      uint4 pk;
      pk.x = pack2bf(acc[0], acc[1]);
      pk.y = pack2bf(acc[2], acc[3]);
      pk.z = pack2bf(acc[4], acc[5]);
      pk.w = pack2bf(acc[6], acc[7]);
      *(uint4*)(smem + VT_OFF + vt_addr(cout, pbase * 2)) = pk;
    }
  }
  __syncthreads();

  // ---------------- Phase C: flash attention + output projection ----------------
  char* stg = smem + STG_OFF + wave * 1024;
  const float* maskrow = mask + (size_t)i * J_DIM;
  float bo_[8];
  #pragma unroll
  for (int n = 0; n < 8; ++n) bo_[n] = b_o[n * 16 + l16];

  for (int qt = wave; qt < 24; qt += 8) {
    const int qrowA = qt * 16 + l16;       // A-fragment row per lane
    const int rbase = qt * 16 + half * 8;  // C-tile row base per lane
    float biasv[8];
    #pragma unroll
    for (int r = 0; r < 8; ++r) biasv[r] = INF_F * (maskrow[rbase + r] - 1.0f);

    // XN A-fragments for this q-tile (all 128 input channels)
    v16bf a_xn[4];
    #pragma unroll
    for (int kc = 0; kc < 4; ++kc)
      a_xn[kc] = ld_frag(smem + XN_OFF + rm_addr(qrowA, kc * 64 + half * 32),
                         smem + XN_OFF + rm_addr(qrowA, kc * 64 + half * 32 + 16));

    v8f oacc[8];
    #pragma unroll
    for (int n = 0; n < 8; ++n) oacc[n] = vzero();

    for (int h = 0; h < N_HEADS; ++h) {
      // ---- Q tile: [16 x 32] = XN @ Wq[:, h*32 : h*32+32] ----
      v8f cq0 = vzero(), cq1 = vzero();
      #pragma unroll
      for (int kc = 0; kc < 4; ++kc) {
        const char* wb0 = (const char*)wqT + ((size_t)((h*2+0)*16 + l16) * 128 + kc*32 + half*16) * 2;
        const char* wb1 = (const char*)wqT + ((size_t)((h*2+1)*16 + l16) * 128 + kc*32 + half*16) * 2;
        cq0 = wmma_bf16(a_xn[kc], ld_frag(wb0, wb0 + 16), cq0);
        cq1 = wmma_bf16(a_xn[kc], ld_frag(wb1, wb1 + 16), cq1);
      }
      // stage Q (scaled by 1/sqrt(d)) and reload as A-fragment
      const float qscale = 0.17677669529663687f;
      #pragma unroll
      for (int r = 0; r < 8; ++r) {
        *(uint16_t*)(stg + st_addr(r + half*8, (0*16 + l16) * 2)) = f2bf(cq0[r] * qscale);
        *(uint16_t*)(stg + st_addr(r + half*8, (1*16 + l16) * 2)) = f2bf(cq1[r] * qscale);
      }
      v16bf qa = ld_frag(stg + st_addr(l16, half * 32),
                         stg + st_addr(l16, half * 32 + 16));

      // ---- flash loop over 12 key chunks of 32 ----
      float m_[8], l_[8];
      #pragma unroll
      for (int r = 0; r < 8; ++r) { m_[r] = -1.0e30f; l_[r] = 0.0f; }
      v8f o0 = vzero(), o1 = vzero();

      for (int kcnk = 0; kcnk < 12; ++kcnk) {
        int kp0 = kcnk * 32 + l16;
        int kp1 = kp0 + 16;
        uint32_t cb = h * 64 + half * 32;
        v16bf kb0 = ld_frag(smem + K_OFF + rm_addr(kp0, cb),
                            smem + K_OFF + rm_addr(kp0, cb + 16));
        v16bf kb1 = ld_frag(smem + K_OFF + rm_addr(kp1, cb),
                            smem + K_OFF + rm_addr(kp1, cb + 16));
        v8f s0 = wmma_bf16(qa, kb0, vzero());
        v8f s1 = wmma_bf16(qa, kb1, vzero());

        // online softmax per row (rows live in 16-lane halves)
        #pragma unroll
        for (int r = 0; r < 8; ++r) {
          float a0 = s0[r] + biasv[r];
          float a1 = s1[r] + biasv[r];
          float mx = redmax16(fmaxf(a0, a1));
          float mn = fmaxf(m_[r], mx);
          float al = __expf(m_[r] - mn);
          float p0 = __expf(a0 - mn);
          float p1 = __expf(a1 - mn);
          l_[r] = l_[r] * al + redsum16(p0 + p1);
          m_[r] = mn;
          o0[r] *= al;
          o1[r] *= al;
          *(uint16_t*)(stg + st_addr(r + half*8, l16 * 2))        = f2bf(p0);
          *(uint16_t*)(stg + st_addr(r + half*8, (16 + l16) * 2)) = f2bf(p1);
        }
        v16bf pa = ld_frag(stg + st_addr(l16, half * 32),
                           stg + st_addr(l16, half * 32 + 16));
        uint32_t pb = kcnk * 64 + half * 32;
        v16bf vb0 = ld_frag(smem + VT_OFF + vt_addr(h * 32 + l16, pb),
                            smem + VT_OFF + vt_addr(h * 32 + l16, pb + 16));
        v16bf vb1 = ld_frag(smem + VT_OFF + vt_addr(h * 32 + 16 + l16, pb),
                            smem + VT_OFF + vt_addr(h * 32 + 16 + l16, pb + 16));
        o0 = wmma_bf16(pa, vb0, o0);
        o1 = wmma_bf16(pa, vb1, o1);
      }
      // normalize, stage hidden[16x32] for this head, fold into output projection
      #pragma unroll
      for (int r = 0; r < 8; ++r) {
        float inv = 1.0f / l_[r];
        *(uint16_t*)(stg + st_addr(r + half*8, l16 * 2))        = f2bf(o0[r] * inv);
        *(uint16_t*)(stg + st_addr(r + half*8, (16 + l16) * 2)) = f2bf(o1[r] * inv);
      }
      v16bf ha = ld_frag(stg + st_addr(l16, half * 32),
                         stg + st_addr(l16, half * 32 + 16));
      #pragma unroll
      for (int n = 0; n < 8; ++n) {
        const char* wb = (const char*)woT + ((size_t)(n * 16 + l16) * 128 + h * 32 + half * 16) * 2;
        oacc[n] = wmma_bf16(ha, ld_frag(wb, wb + 16), oacc[n]);
      }
    }

    // ---- write out: out[i, q, :] = oacc + b_o ----
    #pragma unroll
    for (int n = 0; n < 8; ++n) {
      int col = n * 16 + l16;
      #pragma unroll
      for (int r = 0; r < 8; ++r) {
        out[((size_t)(i * J_DIM + rbase + r)) * C_DIM + col] = oacc[n][r] + bo_[n];
      }
    }
  }
}

// ---------------- host entry ----------------
extern "C" void kernel_launch(void* const* d_in, const int* in_sizes, int n_in,
                              void* d_out, int out_size, void* d_ws, size_t ws_size,
                              hipStream_t stream) {
  (void)in_sizes; (void)n_in; (void)out_size; (void)ws_size;
  const float* x    = (const float*)d_in[0];
  const float* mask = (const float*)d_in[1];
  const float* ln_w = (const float*)d_in[2];
  const float* ln_b = (const float*)d_in[3];
  const float* w_q  = (const float*)d_in[4];
  const float* w_k  = (const float*)d_in[5];
  const float* w_v  = (const float*)d_in[6];
  const float* w_o  = (const float*)d_in[7];
  const float* b_o  = (const float*)d_in[8];
  float*    out = (float*)d_out;
  uint16_t* wsw = (uint16_t*)d_ws;

  ta_prep_weights<<<256, 256, 0, stream>>>(w_q, w_k, w_v, w_o, wsw);

  hipFuncSetAttribute((const void*)ta_fused_kernel,
                      hipFuncAttributeMaxDynamicSharedMemorySize, SMEM_BYTES);
  ta_fused_kernel<<<J_DIM, 256, SMEM_BYTES, stream>>>(x, mask, ln_w, ln_b, b_o, wsw, out);
}
